// CausalAttention_2044404433422
// MI455X (gfx1250) — matrix operand order repaired and checked by reading, not compiled, not run on hardware
//
#include <hip/hip_runtime.h>
#include <hip/hip_bf16.h>
#include <stdint.h>

typedef __bf16 bf16_t;
typedef __attribute__((ext_vector_type(16))) __bf16 v16bf;
typedef __attribute__((ext_vector_type(8)))  __bf16 v8bf;
typedef __attribute__((ext_vector_type(8)))  float  v8f;
typedef __attribute__((ext_vector_type(4)))  float  v4f;

#define P_   2048
#define S_   2048
#define D_   1024
#define H_   16
#define DH_  64
#define DIN_ 8

// ---------------------------------------------------------------------------
// Fragment loaders (CDNA5 WMMA 16x16x32 bf16 VGPR layouts, ISA 7.12.2)
// A (16x32, MxK): lane m=lane&15, g=lane>>4 holds K = [8g..8g+7] and [16+8g..16+8g+7]
// B (32x16, KxN): lane n=lane&15, g=lane>>4 holds K = [8g..8g+7] and [16+8g..16+8g+7] for column n.
// Both operands come from row-major [rows, K] storage (we compute A * B^T),
// so every fragment load is two contiguous 16-byte loads per lane.
// ---------------------------------------------------------------------------
__device__ __forceinline__ v16bf fragA_bf16(const bf16_t* __restrict__ base,
                                            size_t ld, int row0, int k0) {
  const int lane = threadIdx.x & 31;
  const int m = lane & 15, g = lane >> 4;
  const bf16_t* p = base + (size_t)(row0 + m) * ld + (k0 + 8 * g);
  v8bf lo = *(const v8bf*)(p);
  v8bf hi = *(const v8bf*)(p + 16);
  v16bf f;
#pragma unroll
  for (int i = 0; i < 8; ++i) { f[i] = lo[i]; f[i + 8] = hi[i]; }
  return f;
}

__device__ __forceinline__ v16bf fragA_f32(const float* __restrict__ base,
                                           size_t ld, int row0, int k0) {
  const int lane = threadIdx.x & 31;
  const int m = lane & 15, g = lane >> 4;
  const float* p = base + (size_t)(row0 + m) * ld + (k0 + 8 * g);
  v4f a0 = *(const v4f*)(p);
  v4f a1 = *(const v4f*)(p + 4);
  v4f b0 = *(const v4f*)(p + 16);
  v4f b1 = *(const v4f*)(p + 20);
  v16bf f;
#pragma unroll
  for (int i = 0; i < 4; ++i) {
    f[i]      = (bf16_t)a0[i];
    f[i + 4]  = (bf16_t)a1[i];
    f[i + 8]  = (bf16_t)b0[i];
    f[i + 12] = (bf16_t)b1[i];
  }
  return f;
}

// B operand sourced from row-major [N, K] (i.e. B[k][n] = src[n][k]).
// Works for both global and LDS source pointers (compiler picks ds_load/global_load).
__device__ __forceinline__ v16bf fragB_bf16(const bf16_t* base,
                                            size_t ld, int col0, int k0) {
  const int lane = threadIdx.x & 31;
  const int n = lane & 15, g = lane >> 4;
  const bf16_t* p = base + (size_t)(col0 + n) * ld + (k0 + 8 * g);
  v8bf lo = *(const v8bf*)(p);
  v8bf hi = *(const v8bf*)(p + 16);
  v16bf f;
#pragma unroll
  for (int i = 0; i < 8; ++i) { f[i] = lo[i]; f[i + 8] = hi[i]; }
  return f;
}

__device__ __forceinline__ v8f wmma_bf16(v16bf a, v16bf b, v8f c) {
  return __builtin_amdgcn_wmma_f32_16x16x32_bf16(
      /*neg_a=*/false, a, /*neg_b=*/false, b,
      /*c_mod=*/(short)0, c, /*reuse_a=*/false, /*reuse_b=*/false);
}

// ---------------------------------------------------------------------------
// Async global->LDS staging (gfx1250, tracked by ASYNCcnt).
// Each thread issues 16-byte chunks; per-wave async loads complete in order,
// so s_wait_asynccnt N bounds the outstanding tail for double buffering.
// ---------------------------------------------------------------------------
__device__ __forceinline__ void async_b128(uint32_t lds_addr, const void* gaddr) {
  asm volatile("global_load_async_to_lds_b128 %0, %1, off"
               :: "v"(lds_addr), "v"((uint64_t)(uintptr_t)gaddr)
               : "memory");
}
__device__ __forceinline__ void wait_async(int pending) {
  if (pending == 0) asm volatile("s_wait_asynccnt 0x0" ::: "memory");
  else              asm volatile("s_wait_asynccnt 0x2" ::: "memory");
}

// Copy a 64(dh-rows) x 64(k) bf16 slab of vT (row stride S_) into LDS tile
// (row stride 64). 512 16-byte chunks over 256 threads -> 2 async ops/thread.
__device__ __forceinline__ void async_copy_vslab(const bf16_t* __restrict__ vb,
                                                 bf16_t* lds_base, int kk) {
#pragma unroll
  for (int c = threadIdx.x; c < 512; c += 256) {
    const int row = c >> 3;
    const int col = (c & 7) * 8;
    const bf16_t* g = vb + (size_t)row * S_ + kk + col;
    const uint32_t l = (uint32_t)(uintptr_t)(lds_base + row * 64 + col);
    async_b128(l, g);
  }
}

// ---------------------------------------------------------------------------
// Kernel 1: fp32 -> bf16 conversion (inputs + weights into workspace)
// ---------------------------------------------------------------------------
__global__ void k_f32_to_bf16(const float* __restrict__ src,
                              bf16_t* __restrict__ dst, int n) {
  int i = blockIdx.x * blockDim.x + threadIdx.x;
  if (i < n) dst[i] = (bf16_t)src[i];
}

// ---------------------------------------------------------------------------
// Kernel 2: fused QKV projection  C = X * W^T + b, split into heads.
// 2x2 register blocking (32x32 per wave).
// q -> qh[h][p][64], k -> kh[h][s][64], v -> vT[h][64][s] (transposed for PV).
// ---------------------------------------------------------------------------
__global__ void k_qkv_proj(const bf16_t* __restrict__ Qb, const bf16_t* __restrict__ Kb,
                           const bf16_t* __restrict__ Vb,
                           const bf16_t* __restrict__ Wqb, const bf16_t* __restrict__ Wkb,
                           const bf16_t* __restrict__ Wvb,
                           const float* __restrict__ bq, const float* __restrict__ bk,
                           const float* __restrict__ bv,
                           bf16_t* __restrict__ qh, bf16_t* __restrict__ kh,
                           bf16_t* __restrict__ vT) {
  const int wave = blockIdx.x * (blockDim.x >> 5) + (threadIdx.x >> 5);
  const int NB = D_ / 32;                 // 32 column-blocks of 32
  const int per = (P_ / 32) * NB;         // 2048 blocks per matrix
  const int which = wave / per;           // 0=q 1=k 2=v
  const int t = wave % per;
  const int m0 = (t / NB) * 32;
  const int n0 = (t % NB) * 32;

  const bf16_t* X = (which == 0) ? Qb : (which == 1) ? Kb : Vb;
  const bf16_t* W = (which == 0) ? Wqb : (which == 1) ? Wkb : Wvb;
  const float* bias = (which == 0) ? bq : (which == 1) ? bk : bv;

  v8f acc[2][2] = {};
  for (int kk = 0; kk < D_; kk += 32) {
    v16bf a0 = fragA_bf16(X, D_, m0, kk);
    v16bf a1 = fragA_bf16(X, D_, m0 + 16, kk);
    v16bf b0 = fragB_bf16(W, D_, n0, kk);
    v16bf b1 = fragB_bf16(W, D_, n0 + 16, kk);
    acc[0][0] = wmma_bf16(a0, b0, acc[0][0]);
    acc[0][1] = wmma_bf16(a0, b1, acc[0][1]);
    acc[1][0] = wmma_bf16(a1, b0, acc[1][0]);
    acc[1][1] = wmma_bf16(a1, b1, acc[1][1]);
  }

  const int lane = threadIdx.x & 31;
  const int n = lane & 15, g = lane >> 4;
#pragma unroll
  for (int i = 0; i < 2; ++i) {
#pragma unroll
    for (int j = 0; j < 2; ++j) {
      const int d = n0 + 16 * j + n;
      const int h = d >> 6, dh = d & 63;
      const float bb = bias[d];
      const int mbase = m0 + 16 * i;
      if (which == 2) {
        v8bf pk;
#pragma unroll
        for (int r = 0; r < 8; ++r) pk[r] = (bf16_t)(acc[i][j][r] + bb);
        *(v8bf*)(vT + ((size_t)h * DH_ + dh) * S_ + mbase + 8 * g) = pk;
      } else {
        bf16_t* dst = (which == 0) ? qh : kh;
#pragma unroll
        for (int r = 0; r < 8; ++r) {
          const int row = mbase + r + 8 * g;
          dst[((size_t)h * P_ + row) * DH_ + dh] = (bf16_t)(acc[i][j][r] + bb);
        }
      }
    }
  }
}

// ---------------------------------------------------------------------------
// Kernel 3: raw_scores[p,h,s] = gate * (scale * q.k + prev). Per-head GEMM,
// K = 64. 4x4 register blocking (64x64 per wave, 32 WMMAs, 4x operand reuse).
// Gate = (ch_gate[p/256][s/256] >= 0) (STE forward value).
// ---------------------------------------------------------------------------
__global__ void k_scores(const bf16_t* __restrict__ qh, const bf16_t* __restrict__ kh,
                         const float* __restrict__ prev, const float* __restrict__ ch_gate,
                         float* __restrict__ raw) {
  const int wave = blockIdx.x * (blockDim.x >> 5) + (threadIdx.x >> 5);
  const int TS = S_ / 64;                 // 32
  const int perHead = (P_ / 64) * TS;     // 1024
  const int h = wave / perHead;
  const int t = wave % perHead;
  const int p0 = (t / TS) * 64;
  const int s0 = (t % TS) * 64;

  const bf16_t* qb = qh + (size_t)h * P_ * DH_;
  const bf16_t* kb = kh + (size_t)h * S_ * DH_;

  v8f acc[4][4] = {};
#pragma unroll
  for (int kk = 0; kk < DH_; kk += 32) {
    v16bf a[4];
#pragma unroll
    for (int i = 0; i < 4; ++i) a[i] = fragA_bf16(qb, DH_, p0 + 16 * i, kk);
#pragma unroll
    for (int j = 0; j < 4; ++j) {
      v16bf b = fragB_bf16(kb, DH_, s0 + 16 * j, kk);
#pragma unroll
      for (int i = 0; i < 4; ++i) acc[i][j] = wmma_bf16(a[i], b, acc[i][j]);
    }
  }

  const int lane = threadIdx.x & 31;
  const int n = lane & 15, g = lane >> 4;
  const float scale = 0.125f;             // 64^-0.5
#pragma unroll
  for (int j = 0; j < 4; ++j) {
    const int s = s0 + 16 * j + n;
    const int sc = s >> 8;                // s / SEQ_LEN(256)
#pragma unroll
    for (int i = 0; i < 4; ++i) {
#pragma unroll
      for (int r = 0; r < 8; ++r) {
        const int p = p0 + 16 * i + r + 8 * g;
        const int pc = p >> 8;            // p / (P/D_IN) = p/256
        const float gate = (ch_gate[pc * DIN_ + sc] >= 0.f) ? 1.f : 0.f;
        const size_t idx = ((size_t)p * H_ + h) * S_ + s;
        const float v = acc[i][j][r] * scale + prev[idx];
        raw[idx] = __builtin_isfinite(v) ? v * gate : v;
      }
    }
  }
}

// ---------------------------------------------------------------------------
// Kernel 4: masked softmax per (p,h) row of 2048. Max over all s (unmasked,
// per reference), exp, causal mask s<=p, renormalize with clip(sum, 1e-9).
// ---------------------------------------------------------------------------
__global__ void k_softmax(const float* __restrict__ raw, float* __restrict__ attn) {
  __shared__ float red[256];
  const int rowid = blockIdx.x;           // p*H + h
  const int p = rowid / H_;
  const float* rp = raw + (size_t)rowid * S_;
  const int tid = threadIdx.x;

  float x[8];
  float mx = -3.0e38f;
#pragma unroll
  for (int i = 0; i < 8; ++i) {
    x[i] = rp[tid + i * 256];
    mx = fmaxf(mx, x[i]);
  }
  red[tid] = mx; __syncthreads();
  for (int off = 128; off > 0; off >>= 1) {
    if (tid < off) red[tid] = fmaxf(red[tid], red[tid + off]);
    __syncthreads();
  }
  mx = red[0]; __syncthreads();

  float sum = 0.f;
#pragma unroll
  for (int i = 0; i < 8; ++i) {
    const int s = tid + i * 256;
    const float e = (s <= p) ? __expf(x[i] - mx) : 0.f;
    x[i] = e;
    sum += e;
  }
  red[tid] = sum; __syncthreads();
  for (int off = 128; off > 0; off >>= 1) {
    if (tid < off) red[tid] += red[tid + off];
    __syncthreads();
  }
  const float inv = 1.f / fmaxf(red[0], 1e-9f);

  float* op = attn + (size_t)rowid * S_;
#pragma unroll
  for (int i = 0; i < 8; ++i) op[tid + i * 256] = x[i] * inv;
}

// ---------------------------------------------------------------------------
// Kernel 5: PV GEMM, block-cooperative with double-buffered async-LDS staging.
// One block = 8 waves = 8 p-tiles of one head. The per-head vT slab (64x64
// bf16, 8 KB) is staged once per block per K-step with
// global_load_async_to_lds_b128 (ASYNCcnt pipeline), shared by all 8 waves.
// The fp32 attn operand (HBM-resident) is read exactly once.
// ---------------------------------------------------------------------------
__global__ void k_pv(const float* __restrict__ attn, const bf16_t* __restrict__ vT,
                     bf16_t* __restrict__ op) {
  __shared__ __align__(16) bf16_t tile[2][64][64];   // 16 KB double buffer

  const int blk = blockIdx.x;              // 16 heads * 16 p-groups
  const int h  = blk >> 4;
  const int pg = blk & 15;
  const int waveid = threadIdx.x >> 5;
  const int p0 = (pg * 8 + waveid) * 16;

  const float*  ab = attn + (size_t)h * S_;          // row p at ab[p * H*S]
  const bf16_t* vb = vT + (size_t)h * DH_ * S_;

  async_copy_vslab(vb, &tile[0][0][0], 0);

  v8f acc[4] = {};
  const int NSLAB = S_ / 64;               // 32
  for (int t = 0; t < NSLAB; ++t) {
    const int kk = t * 64;
    if (t + 1 < NSLAB) {
      async_copy_vslab(vb, &tile[(t + 1) & 1][0][0], kk + 64);
      wait_async(2);                       // slab t resident (in-order async)
    } else {
      wait_async(0);
    }
    __syncthreads();                       // all waves' chunks visible

    const bf16_t* bt = &tile[t & 1][0][0];
#pragma unroll
    for (int k2 = 0; k2 < 64; k2 += 32) {
      v16bf a = fragA_f32(ab, (size_t)H_ * S_, p0, kk + k2);
#pragma unroll
      for (int j = 0; j < 4; ++j) {
        v16bf b = fragB_bf16(bt, 64, 16 * j, k2);
        acc[j] = wmma_bf16(a, b, acc[j]);
      }
    }
    __syncthreads();                       // buffer may be overwritten next iter
  }

  const int lane = threadIdx.x & 31;
  const int n = lane & 15, g = lane >> 4;
#pragma unroll
  for (int j = 0; j < 4; ++j) {
    const int d = h * DH_ + 16 * j + n;
#pragma unroll
    for (int r = 0; r < 8; ++r) {
      const int row = p0 + r + 8 * g;
      op[(size_t)row * D_ + d] = (bf16_t)acc[j][r];
    }
  }
}

// ---------------------------------------------------------------------------
// Kernel 6: output projection  out = out_pre @ Wo^T + bo (fp32), 2x2 blocked.
// ---------------------------------------------------------------------------
__global__ void k_oproj(const bf16_t* __restrict__ op, const bf16_t* __restrict__ Wob,
                        const float* __restrict__ bo, float* __restrict__ out) {
  const int wave = blockIdx.x * (blockDim.x >> 5) + (threadIdx.x >> 5);
  const int NB = D_ / 32;                 // 32
  const int m0 = (wave / NB) * 32;
  const int n0 = (wave % NB) * 32;

  v8f acc[2][2] = {};
  for (int kk = 0; kk < D_; kk += 32) {
    v16bf a0 = fragA_bf16(op, D_, m0, kk);
    v16bf a1 = fragA_bf16(op, D_, m0 + 16, kk);
    v16bf b0 = fragB_bf16(Wob, D_, n0, kk);
    v16bf b1 = fragB_bf16(Wob, D_, n0 + 16, kk);
    acc[0][0] = wmma_bf16(a0, b0, acc[0][0]);
    acc[0][1] = wmma_bf16(a0, b1, acc[0][1]);
    acc[1][0] = wmma_bf16(a1, b0, acc[1][0]);
    acc[1][1] = wmma_bf16(a1, b1, acc[1][1]);
  }

  const int lane = threadIdx.x & 31;
  const int n = lane & 15, g = lane >> 4;
#pragma unroll
  for (int i = 0; i < 2; ++i) {
#pragma unroll
    for (int j = 0; j < 2; ++j) {
      const int col = n0 + 16 * j + n;
      const float bb = bo[col];
#pragma unroll
      for (int r = 0; r < 8; ++r) {
        const int row = m0 + 16 * i + r + 8 * g;
        out[(size_t)row * D_ + col] = acc[i][j][r] + bb;
      }
    }
  }
}

// ---------------------------------------------------------------------------
// Host launcher
// ---------------------------------------------------------------------------
extern "C" void kernel_launch(void* const* d_in, const int* in_sizes, int n_in,
                              void* d_out, int out_size, void* d_ws, size_t ws_size,
                              hipStream_t stream) {
  (void)in_sizes; (void)n_in; (void)out_size; (void)ws_size;

  const float* Q    = (const float*)d_in[0];
  const float* K    = (const float*)d_in[1];
  const float* V    = (const float*)d_in[2];
  const float* prev = (const float*)d_in[3];
  const float* Wq   = (const float*)d_in[4];
  const float* bq   = (const float*)d_in[5];
  const float* Wk   = (const float*)d_in[6];
  const float* bk   = (const float*)d_in[7];
  const float* Wv   = (const float*)d_in[8];
  const float* bv   = (const float*)d_in[9];
  const float* Wo   = (const float*)d_in[10];
  const float* bo   = (const float*)d_in[11];
  const float* ch_gate = (const float*)d_in[12];
  // d_in[13] = attn_mask (bool) — causal mask computed analytically instead.

  float* out   = (float*)d_out;
  float* attnO = out + (size_t)P_ * D_;                 // 2,097,152
  float* rawO  = attnO + (size_t)P_ * H_ * S_;          // + 67,108,864

  // Workspace layout (bf16 elements).
  const size_t M2 = (size_t)P_ * D_;   // 2M elems
  const size_t M1 = (size_t)D_ * D_;   // 1M elems
  bf16_t* w   = (bf16_t*)d_ws;
  bf16_t* Qb  = w;                 // [P][D]
  bf16_t* Kb  = Qb + M2;           // [S][D]
  bf16_t* Vb  = Kb + M2;           // [S][D]
  bf16_t* Wqb = Vb + M2;           // [D][D]
  bf16_t* Wkb = Wqb + M1;
  bf16_t* Wvb = Wkb + M1;
  bf16_t* Wob = Wvb + M1;
  bf16_t* qh  = Wob + M1;          // [H][P][DH]
  bf16_t* kh  = qh + M2;           // [H][S][DH]
  bf16_t* vT  = kh + M2;           // [H][DH][S]
  bf16_t* opre = vT + M2;          // [P][D]

  // 1) fp32 -> bf16 conversions.
  k_f32_to_bf16<<<8192, 256, 0, stream>>>(Q,  Qb,  (int)M2);
  k_f32_to_bf16<<<8192, 256, 0, stream>>>(K,  Kb,  (int)M2);
  k_f32_to_bf16<<<8192, 256, 0, stream>>>(V,  Vb,  (int)M2);
  k_f32_to_bf16<<<4096, 256, 0, stream>>>(Wq, Wqb, (int)M1);
  k_f32_to_bf16<<<4096, 256, 0, stream>>>(Wk, Wkb, (int)M1);
  k_f32_to_bf16<<<4096, 256, 0, stream>>>(Wv, Wvb, (int)M1);
  k_f32_to_bf16<<<4096, 256, 0, stream>>>(Wo, Wob, (int)M1);

  // 2) QKV projection: 3 * 64 * 32 = 6144 wave-blocks, 8 waves/block -> 768.
  k_qkv_proj<<<768, 256, 0, stream>>>(Qb, Kb, Vb, Wqb, Wkb, Wvb, bq, bk, bv,
                                      qh, kh, vT);

  // 3) Scores: 16 * 32 * 32 = 16384 wave-blocks -> 2048 blocks.
  k_scores<<<2048, 256, 0, stream>>>(qh, kh, prev, ch_gate, rawO);

  // 4) Softmax: one block per (p,h) row -> 32768 blocks.
  k_softmax<<<32768, 256, 0, stream>>>(rawO, attnO);

  // 5) PV: 16 heads * 16 p-groups = 256 blocks (8 p-tiles per block).
  k_pv<<<256, 256, 0, stream>>>(attnO, vT, opre);

  // 6) Output projection: 64 * 32 = 2048 wave-blocks -> 256 blocks.
  k_oproj<<<256, 256, 0, stream>>>(opre, Wob, bo, out);
}
